// GLA2Attention_81741817578093
// MI455X (gfx1250) — compile-verified, matrix operand order checked by reading, not run on hardware
//
#include <hip/hip_runtime.h>

// ---------------- problem constants (from reference) ----------------
#define GB   4
#define GL   2048
#define GD   1024
#define GH   16
#define GDK  64
#define GSCALE 0.125f   // 64^-0.5

// ---------------- CDNA5 WMMA types ----------------
typedef __bf16  v16bf __attribute__((ext_vector_type(16)));
typedef float   v8f   __attribute__((ext_vector_type(8)));

struct __align__(16) U4 { unsigned x, y, z, w; };
struct __align__(16) F4 { float x, y, z, w; };
struct __align__(8)  US4 { unsigned short a, b, c, d; };
struct __align__(32) Frag { U4 lo, hi; };

__device__ __forceinline__ unsigned short f2bf(float f) {
  unsigned u = __builtin_bit_cast(unsigned, f);
  unsigned r = u + 0x7FFFu + ((u >> 16) & 1u);   // round-to-nearest-even
  return (unsigned short)(r >> 16);
}

// =====================================================================
// Elementwise fp32 -> bf16 conversion (hoisted out of the GEMM loop).
// 4 elements / thread; n must be a multiple of 1024 (all our sizes are).
// =====================================================================
__global__ __launch_bounds__(256) void cvt_f32_bf16(
    const float* __restrict__ src, unsigned short* __restrict__ dst) {
  const size_t i = ((size_t)blockIdx.x * 256 + threadIdx.x) * 4;
  F4 f = *reinterpret_cast<const F4*>(src + i);
  US4 o;
  o.a = f2bf(f.x); o.b = f2bf(f.y); o.c = f2bf(f.z); o.d = f2bf(f.w);
  *reinterpret_cast<US4*>(dst + i) = o;
}

// =====================================================================
// GEMM: C[M,N] = A[M,K] (bf16,row-major) x W[K,N] (bf16,row-major), f32 C.
// Per block: 128x64 tile, 128 threads (4 waves), BK=32.
// A tile staged global->LDS with GLOBAL_LOAD_ASYNC_TO_LDS_B128 (ASYNCcnt),
// B tile: vectorized b128 row fetch + b16 transpose-scatter into LDS.
// Math = v_wmma_f32_16x16x32_bf16 with fp32 accumulators (2x4 per wave).
// =====================================================================
__global__ __launch_bounds__(128) void gemm_bf16_wmma(
    const unsigned short* __restrict__ A, const unsigned short* __restrict__ W,
    float* __restrict__ C, int M, int N, int K) {
  const int tid   = threadIdx.x;
  const int wave  = tid >> 5;
  const int lane  = tid & 31;
  const int blockM = blockIdx.y * 128;
  const int blockN = blockIdx.x * 64;

  __shared__ __align__(16) unsigned short As[128 * 32];  // [m][k] bf16
  __shared__ __align__(16) unsigned short Bs[64 * 32];   // [n][k] bf16 (transposed)

  const v8f vzero = {0.f, 0.f, 0.f, 0.f, 0.f, 0.f, 0.f, 0.f};
  v8f acc[2][4];
#pragma unroll
  for (int i = 0; i < 2; ++i)
#pragma unroll
    for (int j = 0; j < 4; ++j) acc[i][j] = vzero;

  const int k0sel = (lane >> 4) << 3;   // 0 for lanes 0-15, 8 for lanes 16-31
  const int mrow  = lane & 15;
  const size_t arow = (size_t)(blockM + tid) * (size_t)K;   // A row (elements)

  const unsigned long long abase = (unsigned long long)(uintptr_t)A;
  const unsigned lds_a = (unsigned)(uintptr_t)(&As[tid * 32]); // LDS byte addr

  // --- precomputed per-thread B-staging geometry (all 32-bit math) ---
  // 256 segments of 8 consecutive bf16; thread handles segments tid, tid+128.
  int wseg[2], lseg[2];
#pragma unroll
  for (int i2 = 0; i2 < 2; ++i2) {
    int s  = tid + i2 * 128;          // 0..255
    int kk = s >> 3;                  // 0..31 (B-tile row)
    int n0 = (s & 7) << 3;            // 0,8,..,56 (col octet)
    wseg[i2] = kk * N + blockN + n0;  // W element offset sans kb*N
    lseg[i2] = n0 * 32 + kk;          // Bs element offset, stride 32 per n
  }

  for (int kb = 0; kb < K; kb += 32) {
    // ---- stage A tile via async DMA to LDS: 64B per thread-row ----
    {
      const unsigned goff = (unsigned)((arow + (size_t)kb) * 2u); // byte offset
      asm volatile("global_load_async_to_lds_b128 %0, %1, %2 offset:0"
                   :: "v"(lds_a), "v"(goff), "s"(abase) : "memory");
      asm volatile("global_load_async_to_lds_b128 %0, %1, %2 offset:16"
                   :: "v"(lds_a), "v"(goff), "s"(abase) : "memory");
      asm volatile("global_load_async_to_lds_b128 %0, %1, %2 offset:32"
                   :: "v"(lds_a), "v"(goff), "s"(abase) : "memory");
      asm volatile("global_load_async_to_lds_b128 %0, %1, %2 offset:48"
                   :: "v"(lds_a), "v"(goff), "s"(abase) : "memory");
    }
    // ---- stage B tile: two b128 row fetches, then transpose-scatter ----
    {
      const int kbn = kb * N;
      U4 d0 = *reinterpret_cast<const U4*>(W + (wseg[0] + kbn));
      U4 d1 = *reinterpret_cast<const U4*>(W + (wseg[1] + kbn));
      unsigned short* b0 = &Bs[lseg[0]];
      unsigned short* b1 = &Bs[lseg[1]];
      b0[0 * 32] = (unsigned short)d0.x; b0[1 * 32] = (unsigned short)(d0.x >> 16);
      b0[2 * 32] = (unsigned short)d0.y; b0[3 * 32] = (unsigned short)(d0.y >> 16);
      b0[4 * 32] = (unsigned short)d0.z; b0[5 * 32] = (unsigned short)(d0.z >> 16);
      b0[6 * 32] = (unsigned short)d0.w; b0[7 * 32] = (unsigned short)(d0.w >> 16);
      b1[0 * 32] = (unsigned short)d1.x; b1[1 * 32] = (unsigned short)(d1.x >> 16);
      b1[2 * 32] = (unsigned short)d1.y; b1[3 * 32] = (unsigned short)(d1.y >> 16);
      b1[4 * 32] = (unsigned short)d1.z; b1[5 * 32] = (unsigned short)(d1.z >> 16);
      b1[6 * 32] = (unsigned short)d1.w; b1[7 * 32] = (unsigned short)(d1.w >> 16);
    }
    if (kb + 32 < K)
      __builtin_prefetch(A + arow + kb + 32, 0, 1);   // global_prefetch_b8
    asm volatile("s_wait_asynccnt 0x0" ::: "memory");
    __syncthreads();

    // ---- build fragments per ISA 16-bit 16x32 layout ----
    v16bf afrag[2], bfrag[4];
#pragma unroll
    for (int i = 0; i < 2; ++i) {
      int m = wave * 32 + i * 16 + mrow;
      const U4* p = reinterpret_cast<const U4*>(&As[m * 32 + k0sel]);
      Frag fr; fr.lo = p[0]; fr.hi = p[2];   // K k0..k0+7 and k0+16..k0+23
      afrag[i] = __builtin_bit_cast(v16bf, fr);
    }
#pragma unroll
    for (int j = 0; j < 4; ++j) {
      int n = j * 16 + mrow;
      const U4* p = reinterpret_cast<const U4*>(&Bs[n * 32 + k0sel]);
      Frag fr; fr.lo = p[0]; fr.hi = p[2];
      bfrag[j] = __builtin_bit_cast(v16bf, fr);
    }
#pragma unroll
    for (int i = 0; i < 2; ++i)
#pragma unroll
      for (int j = 0; j < 4; ++j)
        acc[i][j] = __builtin_amdgcn_wmma_f32_16x16x32_bf16(
            false, afrag[i], false, bfrag[j], (short)0, acc[i][j], false, false);
    __syncthreads();
  }

  // ---- epilogue: C/D layout: lane gives N=lane%16, M = r + 8*(lane/16) ----
  const int rbase = blockM + wave * 32 + ((lane >> 4) << 3);
  const int cbase = blockN + (lane & 15);
#pragma unroll
  for (int i = 0; i < 2; ++i)
#pragma unroll
    for (int j = 0; j < 4; ++j)
#pragma unroll
      for (int r = 0; r < 8; ++r) {
        int row = rbase + i * 16 + r;
        int col = cbase + j * 16;
        C[(size_t)row * (size_t)N + col] = acc[i][j][r];
      }
}

// =====================================================================
// Row softmax helpers (rows of length 1024, 256 threads, 4 elems/thread)
// =====================================================================
__device__ __forceinline__ float blk_reduce_max(float v, float* red) {
  int t = threadIdx.x;
  red[t] = v; __syncthreads();
  for (int s = 128; s > 0; s >>= 1) {
    if (t < s) red[t] = fmaxf(red[t], red[t + s]);
    __syncthreads();
  }
  float r = red[0]; __syncthreads();
  return r;
}
__device__ __forceinline__ float blk_reduce_sum(float v, float* red) {
  int t = threadIdx.x;
  red[t] = v; __syncthreads();
  for (int s = 128; s > 0; s >>= 1) {
    if (t < s) red[t] = red[t] + red[t + s];
    __syncthreads();
  }
  float r = red[0]; __syncthreads();
  return r;
}

// q = softmax(q_logits), in place
__global__ __launch_bounds__(256) void softmax_rows(float* __restrict__ X) {
  __shared__ float red[256];
  const size_t base = (size_t)blockIdx.x * GD;
  const int t = threadIdx.x;
  float x[4];
#pragma unroll
  for (int u = 0; u < 4; ++u) x[u] = X[base + t + u * 256];
  float m = fmaxf(fmaxf(x[0], x[1]), fmaxf(x[2], x[3]));
  m = blk_reduce_max(m, red);
  float e[4], ls = 0.f;
#pragma unroll
  for (int u = 0; u < 4; ++u) { e[u] = __expf(x[u] - m); ls += e[u]; }
  float s = blk_reduce_sum(ls, red);
  float inv = 1.0f / s;
#pragma unroll
  for (int u = 0; u < 4; ++u) X[base + t + u * 256] = e[u] * inv;
}

// k = 1 - softmax(f_logits) (in place over F); g = log_softmax(f_logits)
__global__ __launch_bounds__(256) void softmax_fg(float* __restrict__ F,
                                                  float* __restrict__ G) {
  __shared__ float red[256];
  const size_t base = (size_t)blockIdx.x * GD;
  const int t = threadIdx.x;
  float x[4];
#pragma unroll
  for (int u = 0; u < 4; ++u) x[u] = F[base + t + u * 256];
  float m = fmaxf(fmaxf(x[0], x[1]), fmaxf(x[2], x[3]));
  m = blk_reduce_max(m, red);
  float e[4], ls = 0.f;
#pragma unroll
  for (int u = 0; u < 4; ++u) { e[u] = __expf(x[u] - m); ls += e[u]; }
  float s = blk_reduce_sum(ls, red);
  float inv = 1.0f / s;
  float logs = __logf(s);
#pragma unroll
  for (int u = 0; u < 4; ++u) {
    F[base + t + u * 256] = 1.0f - e[u] * inv;        // k
    G[base + t + u * 256] = (x[u] - m) - logs;        // log_softmax
  }
}

// =====================================================================
// GLA recurrence: one block per (b,h); 64 threads, each owns one value
// column of the 64x64 state, held in 64 VGPRs. Sequential over L.
//   S = S * exp(g)[:,None] + k[:,None]*v[None,:] ; o = (q*scale) @ S
// =====================================================================
__global__ __launch_bounds__(64) void gla_scan(
    const float* __restrict__ Q, const float* __restrict__ Kb,
    const float* __restrict__ G, const float* __restrict__ V,
    float* __restrict__ O) {
  const int b = blockIdx.x / GH;
  const int h = blockIdx.x % GH;
  const int tid = threadIdx.x;          // value column 0..63

  __shared__ float sq[GDK], sk[GDK], sg[GDK];
  float S[GDK];
#pragma unroll
  for (int kk = 0; kk < GDK; ++kk) S[kk] = 0.f;

  for (int t = 0; t < GL; ++t) {
    const size_t base = ((size_t)b * GL + t) * GD + h * GDK;
    sq[tid] = Q[base + tid];
    sk[tid] = Kb[base + tid];
    sg[tid] = G[base + tid];
    const float vv = V[base + tid];
    __syncthreads();
    float acc = 0.f;
#pragma unroll
    for (int kk = 0; kk < GDK; ++kk) {
      float sn = S[kk] * __expf(sg[kk]) + sk[kk] * vv;
      S[kk] = sn;
      acc += sq[kk] * sn;
    }
    O[base + tid] = acc * GSCALE;
    __syncthreads();
  }
}

// =====================================================================
// Launch pipeline
// =====================================================================
extern "C" void kernel_launch(void* const* d_in, const int* in_sizes, int n_in,
                              void* d_out, int out_size, void* d_ws, size_t ws_size,
                              hipStream_t stream) {
  (void)in_sizes; (void)n_in; (void)out_size; (void)ws_size;
  const float* hs = (const float*)d_in[0];
  const float* Wq = (const float*)d_in[1];
  const float* Wf = (const float*)d_in[2];
  const float* Wi = (const float*)d_in[3];
  const float* Wo = (const float*)d_in[4];
  float* out = (float*)d_out;

  const size_t NEL = (size_t)GB * GL * GD;   // 8,388,608
  const size_t WEL = (size_t)GD * GD;        // 1,048,576
  float* ws = (float*)d_ws;
  float* Qb = ws + 0 * NEL;   // q logits -> q
  float* Kf = ws + 1 * NEL;   // f logits -> k
  float* Gb = ws + 2 * NEL;   // log-softmax gates
  float* Vb = ws + 3 * NEL;   // values
  float* Ob = ws + 4 * NEL;   // scan output

  unsigned short* bfh = (unsigned short*)(ws + 5 * NEL);
  unsigned short* hs16 = bfh;                 // 8M
  unsigned short* wq16 = bfh + NEL;           // 1M
  unsigned short* wf16 = wq16 + WEL;
  unsigned short* wi16 = wf16 + WEL;
  unsigned short* wo16 = wi16 + WEL;
  unsigned short* ob16 = wo16 + WEL;          // 8M

  const int M = GB * GL, N = GD, K = GD;     // 8192 x 1024 x 1024
  dim3 gGrid(N / 64, M / 128);               // (16, 64)
  dim3 gBlk(128);

  // one-time bf16 conversions (bandwidth-trivial; keeps GEMM loop math-free)
  cvt_f32_bf16<<<dim3((unsigned)(NEL / 1024)), dim3(256), 0, stream>>>(hs, hs16);
  cvt_f32_bf16<<<dim3((unsigned)(WEL / 1024)), dim3(256), 0, stream>>>(Wq, wq16);
  cvt_f32_bf16<<<dim3((unsigned)(WEL / 1024)), dim3(256), 0, stream>>>(Wf, wf16);
  cvt_f32_bf16<<<dim3((unsigned)(WEL / 1024)), dim3(256), 0, stream>>>(Wi, wi16);
  cvt_f32_bf16<<<dim3((unsigned)(WEL / 1024)), dim3(256), 0, stream>>>(Wo, wo16);

  // projections (bf16 WMMA, fp32 accumulate)
  gemm_bf16_wmma<<<gGrid, gBlk, 0, stream>>>(hs16, wq16, Qb, M, N, K);
  gemm_bf16_wmma<<<gGrid, gBlk, 0, stream>>>(hs16, wf16, Kf, M, N, K);
  gemm_bf16_wmma<<<gGrid, gBlk, 0, stream>>>(hs16, wi16, Vb, M, N, K);

  // gates over full feature dim D
  softmax_rows<<<dim3(GB * GL), dim3(256), 0, stream>>>(Qb);
  softmax_fg  <<<dim3(GB * GL), dim3(256), 0, stream>>>(Kf, Gb);

  // sequential gated recurrence, state in VGPRs
  gla_scan<<<dim3(GB * GH), dim3(64), 0, stream>>>(Qb, Kf, Gb, Vb, Ob);

  // output projection
  cvt_f32_bf16<<<dim3((unsigned)(NEL / 1024)), dim3(256), 0, stream>>>(Ob, ob16);
  gemm_bf16_wmma<<<gGrid, gBlk, 0, stream>>>(ob16, wo16, out, M, N, K);
}
